// ExpertRouter_13804024889902
// MI455X (gfx1250) — compile-verified, matrix operand order checked by reading, not run on hardware
//
#include <hip/hip_runtime.h>
#include <hip/hip_bf16.h>
#include <math.h>

// ---------------------------------------------------------------------------
// MoE expert router for MI455X (gfx1250, wave32).
//   x[16384,2048] fp32 -> silu(x@W1+b1) -> @W2+b2 -> logits[16384,64]
//   -> top2/softmax weights, aux load-balance loss, capacity drop mask.
//
// GEMM1 (68.7 GFLOP, ~97% of all work) runs on v_wmma_f32_16x16x32_bf16.
// Tiling: 32 tokens per WG (2 M-tiles), 8 waves; each wave holds 2x8 fp32
// accumulator tiles so every B-fragment (global/L2) feeds 2 wmma and every
// A-fragment (LDS) is amortized over 8 wmma. LDS rows padded by 16 B so the
// strided ds_load_b128 A-fetches are bank-conflict-free.
// ---------------------------------------------------------------------------

typedef __bf16 bf16;
typedef __attribute__((ext_vector_type(16))) __bf16 v16bf;
typedef __attribute__((ext_vector_type(8)))  __bf16 v8bf;
typedef __attribute__((ext_vector_type(8)))  float  v8f;

#define B_TOK   16384
#define D_IN    2048
#define H_DIM   1024
#define N_EXP   64
#define TOPK    2
#define CAPACITY 640          // ceil(16384*2/64 * 1.25)

#define M_WG    32            // tokens per workgroup (2 x 16-row M tiles)
#define XPITCH  (D_IN + 8)    // LDS row pitch (bf16) -> 4112 B, bank stride 4
#define HPITCH  (H_DIM + 8)   // LDS row pitch (bf16) -> 2064 B, bank stride 4

union AFrag { v16bf v; v8bf h[2]; };

// K index inside a 32-wide chunk for (lane-half, element) per ISA 7.12.2
// (16-bit A-matrix 16x32): lanes 0-15 -> K {0..7, 16..23}, lanes 16-31 ->
// K {8..15, 24..31}. B fragments are packed with the mirrored mapping so a
// lane's 16 bf16 values are contiguous (32 bytes).
__device__ __forceinline__ int k_in_chunk(int half, int e) {
  return (e < 8) ? (half * 8 + e) : (16 + half * 8 + (e - 8));
}

__device__ __forceinline__ float fast_sigmoid(float v) {
  return __builtin_amdgcn_rcpf(1.0f + __expf(-v));   // v_rcp_f32 + v_exp_f32
}

// ---------------------------------------------------------------------------
// Kernel 1: pack W1/W2 fp32 -> bf16 in WMMA B-fragment order; zero accums.
// w1p index: (((j*64 + c)*32 + lane)*16 + e),  j = H/16 tile, c = K/32 chunk
// w2p index: (((j*32 + c)*32 + lane)*16 + e),  j = E/16 tile, c = K/32 chunk
// ---------------------------------------------------------------------------
__global__ __launch_bounds__(256) void moe_pack_weights(
    const float* __restrict__ W1, const float* __restrict__ W2,
    bf16* __restrict__ w1p, bf16* __restrict__ w2p,
    float* __restrict__ probs_sum, int* __restrict__ freq_cnt)
{
  const int gsz = gridDim.x * blockDim.x;
  const int gid = blockIdx.x * blockDim.x + threadIdx.x;
  if (gid < N_EXP) { probs_sum[gid] = 0.0f; freq_cnt[gid] = 0; }

  const int total1 = D_IN * H_DIM;
  for (int p = gid; p < total1; p += gsz) {
    const int e    = p & 15;
    const int lane = (p >> 4) & 31;
    const int c    = (p >> 9) & 63;
    const int j    = p >> 15;
    const int k    = c * 32 + k_in_chunk(lane >> 4, e);
    const int n    = j * 16 + (lane & 15);
    w1p[p] = (bf16)W1[(size_t)k * H_DIM + n];
  }
  const int total2 = H_DIM * N_EXP;
  for (int p = gid; p < total2; p += gsz) {
    const int e    = p & 15;
    const int lane = (p >> 4) & 31;
    const int c    = (p >> 9) & 31;
    const int j    = p >> 14;
    const int k    = c * 32 + k_in_chunk(lane >> 4, e);
    const int n    = j * 16 + (lane & 15);
    w2p[p] = (bf16)W2[(size_t)k * N_EXP + n];
  }
}

// ---------------------------------------------------------------------------
// Kernel 2: fused gate MLP. One WG (8 waves) per 32-token tile.
//   LDS: sX = 32 x XPITCH bf16 (128.5 KB), sH = 32 x HPITCH bf16 (64.5 KB).
// ---------------------------------------------------------------------------
__global__ __launch_bounds__(256) void moe_gate_mlp(
    const float* __restrict__ x, const bf16* __restrict__ w1p,
    const bf16* __restrict__ w2p, const float* __restrict__ b1,
    const float* __restrict__ b2, float* __restrict__ logits)
{
  extern __shared__ __align__(16) char smem[];
  bf16* sX = (bf16*)smem;                                        // [32][XPITCH]
  bf16* sH = (bf16*)(smem + (size_t)M_WG * XPITCH * sizeof(bf16)); // [32][HPITCH]

  const int tid  = threadIdx.x;
  const int wave = tid >> 5;
  const int lane = tid & 31;
  const int m0   = blockIdx.x * M_WG;

  // Stage x tile fp32 -> bf16 into LDS, float4-vectorized.
  for (int i = tid; i < M_WG * (D_IN / 4); i += 256) {
    const int r = i >> 9;               // D_IN/4 == 512
    const int c = (i & 511) * 4;
    const float4 v = *(const float4*)(x + (size_t)(m0 + r) * D_IN + c);
    bf16* dst = sX + r * XPITCH + c;
    dst[0] = (bf16)v.x; dst[1] = (bf16)v.y;
    dst[2] = (bf16)v.z; dst[3] = (bf16)v.w;
  }
  __syncthreads();

  const int mrow  = lane & 15;   // A-matrix row for this lane
  const int khalf = lane >> 4;   // which K-half this lane carries

  // ---- GEMM1: h = silu(x @ W1 + b1)
  // Wave w owns H-tiles j = w*8 .. w*8+7 for both M-tiles.
  {
    const v8f zero8 = {0.f, 0.f, 0.f, 0.f, 0.f, 0.f, 0.f, 0.f};
    v8f acc[16];                       // [jj][mt], 128 VGPRs of accumulators
    #pragma unroll
    for (int q = 0; q < 16; ++q) acc[q] = zero8;

    const bf16* paBase = sX + mrow * XPITCH + khalf * 8;
    #pragma unroll 2
    for (int c = 0; c < D_IN / 32; ++c) {
      AFrag a0, a1;                    // A fragments for the two M tiles
      const bf16* pa = paBase + c * 32;
      a0.h[0] = *(const v8bf*)pa;                       // K k0..k0+7
      a0.h[1] = *(const v8bf*)(pa + 16);                // K k0+16..k0+23
      a1.h[0] = *(const v8bf*)(pa + 16 * XPITCH);
      a1.h[1] = *(const v8bf*)(pa + 16 * XPITCH + 16);
      #pragma unroll
      for (int jj = 0; jj < 8; ++jj) {
        const bf16* bp = w1p + (((size_t)(wave * 8 + jj) * 64 + c) * 32 + lane) * 16;
        const v16bf bmat = *(const v16bf*)bp;
        __builtin_prefetch(bp + 512, 0, 1);             // next k-chunk, same j
        acc[jj * 2 + 0] = __builtin_amdgcn_wmma_f32_16x16x32_bf16(
            false, a0.v, false, bmat, (short)0, acc[jj * 2 + 0], false, false);
        acc[jj * 2 + 1] = __builtin_amdgcn_wmma_f32_16x16x32_bf16(
            false, a1.v, false, bmat, (short)0, acc[jj * 2 + 1], false, false);
      }
    }

    // bias + SiLU, write bf16 h tiles to LDS (C layout: VGPR r -> M = r+8*khalf)
    #pragma unroll
    for (int jj = 0; jj < 8; ++jj) {
      const int ncol = (wave * 8 + jj) * 16 + mrow;
      const float bias = b1[ncol];
      #pragma unroll
      for (int mt = 0; mt < 2; ++mt) {
        const v8f a = acc[jj * 2 + mt];
        #pragma unroll
        for (int r = 0; r < 8; ++r) {
          const float v = a[r] + bias;
          sH[(mt * 16 + r + khalf * 8) * HPITCH + ncol] = (bf16)(v * fast_sigmoid(v));
        }
      }
    }
  }
  __syncthreads();

  // ---- GEMM2: logits = h @ W2 + b2. 8 waves = 2 M-tiles x 4 expert tiles.
  {
    const int mt = wave >> 2;          // which 16-token tile
    const int j2 = wave & 3;           // which 16-expert tile
    v8f acc = {0.f, 0.f, 0.f, 0.f, 0.f, 0.f, 0.f, 0.f};
    const bf16* paBase = sH + (mt * 16 + mrow) * HPITCH + khalf * 8;
    const bf16* bbase  = w2p + ((size_t)j2 * 32 * 32 + lane) * 16;
    #pragma unroll 4
    for (int c = 0; c < H_DIM / 32; ++c) {
      AFrag a;
      const bf16* pa = paBase + c * 32;
      a.h[0] = *(const v8bf*)pa;
      a.h[1] = *(const v8bf*)(pa + 16);
      const v16bf bmat = *(const v16bf*)(bbase + (size_t)c * (32 * 16));
      acc = __builtin_amdgcn_wmma_f32_16x16x32_bf16(
          false, a.v, false, bmat, (short)0, acc, false, false);
    }
    const int ncol = j2 * 16 + mrow;
    const float bias = b2[ncol];
    #pragma unroll
    for (int r = 0; r < 8; ++r)
      logits[(size_t)(m0 + mt * 16 + r + khalf * 8) * N_EXP + ncol] = acc[r] + bias;
  }
}

// ---------------------------------------------------------------------------
// Kernel 3: per-token top-2 + softmax weights; block-local LDS reductions
// (ds_add_f32 / ds int atomics) of softmax prob sums + argmax histogram.
// Logits are loaded once into registers and reused across all passes.
// ---------------------------------------------------------------------------
__global__ __launch_bounds__(256) void moe_routing(
    const float* __restrict__ logits, float* __restrict__ out,
    int* __restrict__ topidx, float* __restrict__ probs_sum,
    int* __restrict__ freq_cnt)
{
  __shared__ float ps[N_EXP];
  __shared__ int   fc[N_EXP];
  const int tid = threadIdx.x;
  if (tid < N_EXP) { ps[tid] = 0.0f; fc[tid] = 0; }
  __syncthreads();

  const int t = blockIdx.x * blockDim.x + tid;
  const float* lg = logits + (size_t)t * N_EXP;

  float lv[N_EXP];
  #pragma unroll
  for (int e = 0; e < N_EXP; ++e) lv[e] = lg[e];

  float m1 = -INFINITY, m2 = -INFINITY;
  int i1 = 0, i2 = 0;
  #pragma unroll
  for (int e = 0; e < N_EXP; ++e) {
    const float v = lv[e];
    if (v > m1)      { m2 = m1; i2 = i1; m1 = v; i1 = e; }
    else if (v > m2) { m2 = v;  i2 = e; }
  }
  // softmax over the top-2 values (descending order, matches top_k)
  const float w0 = fast_sigmoid(m1 - m2);   // 1/(1+exp(m2-m1))
  out[t * 2 + 0] = w0;
  out[t * 2 + 1] = 1.0f - w0;
  out[B_TOK * 2 + t * 2 + 0] = (float)i1;   // top_idx as float output
  out[B_TOK * 2 + t * 2 + 1] = (float)i2;
  topidx[t * 2 + 0] = i1;                   // int copy for the drop kernel
  topidx[t * 2 + 1] = i2;

  // full softmax contribution for the aux loss (exp computed once per expert)
  float sum = 0.0f;
  #pragma unroll
  for (int e = 0; e < N_EXP; ++e) { lv[e] = __expf(lv[e] - m1); sum += lv[e]; }
  const float inv = __builtin_amdgcn_rcpf(sum);
  #pragma unroll
  for (int e = 0; e < N_EXP; ++e)
    atomicAdd(&ps[e], lv[e] * inv);          // ds_add_f32
  atomicAdd(&fc[i1], 1);
  __syncthreads();
  if (tid < N_EXP) {
    atomicAdd(&probs_sum[tid], ps[tid]);
    atomicAdd(&freq_cnt[tid], fc[tid]);
  }
}

// ---------------------------------------------------------------------------
// Kernel 4: aux loss finalize + order-exact k-major capacity drop scan.
// Single WG: LDS expert counters carried across 256-item chunks; within a
// chunk each thread ranks itself against earlier threads (LDS compare loop),
// reproducing the sequential k-major / batch-minor drop semantics exactly.
// ---------------------------------------------------------------------------
__global__ __launch_bounds__(256) void moe_aux_drop(
    const float* __restrict__ probs_sum, const int* __restrict__ freq_cnt,
    const int* __restrict__ topidx, float* __restrict__ out)
{
  __shared__ float red[N_EXP];
  __shared__ int   cnt[N_EXP];
  __shared__ int   eid[256];
  const int tid = threadIdx.x;

  if (tid < N_EXP) {
    const float f  = (float)freq_cnt[tid] / (float)B_TOK;   // top1 frequency
    const float pm = probs_sum[tid] / (float)B_TOK;         // mean prob
    red[tid] = f * pm;
    cnt[tid] = 0;
  }
  __syncthreads();
  if (tid == 0) {
    float s = 0.0f;
    for (int e = 0; e < N_EXP; ++e) s += red[e];
    out[B_TOK * TOPK * 2] = (float)N_EXP * s;               // aux_loss
  }

  float* drop = out + B_TOK * TOPK * 2 + 1;
  for (int it = 0; it < (B_TOK * TOPK) / 256; ++it) {
    const int i = it * 256 + tid;       // flat k-major index
    const int k = i >> 14;              // B_TOK == 1<<14
    const int b = i & (B_TOK - 1);
    const int e = topidx[b * TOPK + k];
    eid[tid] = e;
    __syncthreads();
    int prior = cnt[e];                 // count before this chunk
    for (int j = 0; j < tid; ++j)
      if (eid[j] == e) ++prior;         // + earlier items within this chunk
    drop[b * TOPK + k] = (prior >= CAPACITY) ? 1.0f : 0.0f;
    __syncthreads();
    atomicAdd(&cnt[e], 1);              // commit this chunk's counts
    __syncthreads();
  }
}

// ---------------------------------------------------------------------------
extern "C" void kernel_launch(void* const* d_in, const int* in_sizes, int n_in,
                              void* d_out, int out_size, void* d_ws, size_t ws_size,
                              hipStream_t stream)
{
  (void)in_sizes; (void)n_in; (void)out_size; (void)ws_size;
  const float* x  = (const float*)d_in[0];
  const float* W1 = (const float*)d_in[1];
  const float* b1 = (const float*)d_in[2];
  const float* W2 = (const float*)d_in[3];
  const float* b2 = (const float*)d_in[4];
  float* out = (float*)d_out;

  // workspace layout (~8.6 MB total)
  char* ws = (char*)d_ws;
  bf16*  w1p       = (bf16*)(ws);                                   // 4 MiB
  bf16*  w2p       = (bf16*)(ws + (size_t)4 * 1024 * 1024);         // 128 KiB
  float* logits    = (float*)(ws + (size_t)4 * 1024 * 1024 + 128 * 1024); // 4 MiB
  float* probs_sum = (float*)(ws + (size_t)8 * 1024 * 1024 + 128 * 1024); // 256 B
  int*   freq_cnt  = (int*)  (ws + (size_t)8 * 1024 * 1024 + 128 * 1024 + 256);
  int*   topidx    = (int*)  (ws + (size_t)8 * 1024 * 1024 + 128 * 1024 + 512); // 128 KiB

  const size_t smem = (size_t)(M_WG * XPITCH + M_WG * HPITCH) * sizeof(bf16); // ~193 KB
  // gfx1250 WGPs have 320 KB LDS; raise the dynamic-LDS cap. Idempotent
  // host-side call, done unconditionally every launch (no static state),
  // safe under graph capture.
  (void)hipFuncSetAttribute((const void*)moe_gate_mlp,
                            hipFuncAttributeMaxDynamicSharedMemorySize,
                            (int)smem);

  moe_pack_weights<<<512, 256, 0, stream>>>(W1, W2, w1p, w2p, probs_sum, freq_cnt);
  moe_gate_mlp<<<B_TOK / M_WG, 256, smem, stream>>>(x, w1p, w2p, b1, b2, logits);
  moe_routing<<<B_TOK / 256, 256, 0, stream>>>(logits, out, topidx,
                                               probs_sum, freq_cnt);
  moe_aux_drop<<<1, 256, 0, stream>>>(probs_sum, freq_cnt, topidx, out);
}
